// CustomGraphNet_6880537608483
// MI455X (gfx1250) — matrix operand order repaired and checked by reading, not compiled, not run on hardware
//
#include <hip/hip_runtime.h>
#include <hip/hip_bf16.h>
#include <math.h>

static constexpr int NN  = 10000;   // nodes
static constexpr int NE  = 120000;  // edges
static constexpr int CC  = 128;     // channels
static constexpr int HH  = 4;       // heads
static constexpr int DD  = 768;     // cond dim
static constexpr int HCC = 512;     // H*C

typedef __bf16 bf16x16 __attribute__((ext_vector_type(16)));
typedef float  f32x8   __attribute__((ext_vector_type(8)));

union FragAB { bf16x16 v; uint4 q[2]; };

// ---------------------------------------------------------------------------
// WMMA GEMM: OUT[M x Kout] = A_bf16[M x Kin] @ (WT_bf16[Kout x Kin])^T + bias
// One wave computes a 16x32 tile: 1 A-frag -> 2 wmma per 32-wide K step.
// blockDim = (32,4), grid = (M/16, Kout/128).
// ---------------------------------------------------------------------------
__global__ void k_gemm_bf16(const __hip_bfloat16* __restrict__ A,
                            const __hip_bfloat16* __restrict__ WT,
                            const float* __restrict__ bias,
                            float* __restrict__ OUT,
                            int Kin, int Kout)
{
    const int lane  = threadIdx.x;
    const int half  = lane >> 4;       // 0 or 1
    const int l16   = lane & 15;
    const int mBase = blockIdx.x * 16;
    const int nBase = (blockIdx.y * blockDim.y + threadIdx.y) * 32;

    const __hip_bfloat16* arow = A  + (size_t)(mBase + l16) * Kin;
    const __hip_bfloat16* w0   = WT + (size_t)(nBase + l16) * Kin;
    const __hip_bfloat16* w1   = w0 + (size_t)16 * Kin;

    f32x8 acc0 = {};
    f32x8 acc1 = {};
    for (int kk = 0; kk < Kin; kk += 32) {
        FragAB a, b0, b1;
        // A 16x32 (contiguous k order, same as B): elems j -> K = kk + half*16 + j
        a.q[0] = *(const uint4*)(arow + kk + half * 16);
        a.q[1] = *(const uint4*)(arow + kk + half * 16 + 8);
        // B 32x16: lane%16 = column, elems j -> K = kk + half*16 + j
        const int kb = kk + half * 16;
        b0.q[0] = *(const uint4*)(w0 + kb);
        b0.q[1] = *(const uint4*)(w0 + kb + 8);
        b1.q[0] = *(const uint4*)(w1 + kb);
        b1.q[1] = *(const uint4*)(w1 + kb + 8);
        acc0 = __builtin_amdgcn_wmma_f32_16x16x32_bf16(false, a.v, false, b0.v,
                                                       (short)0, acc0, false, false);
        acc1 = __builtin_amdgcn_wmma_f32_16x16x32_bf16(false, a.v, false, b1.v,
                                                       (short)0, acc1, false, false);
    }

    const int n0 = nBase + l16;
    const int n1 = n0 + 16;
    const float bi0 = bias ? bias[n0] : 0.0f;
    const float bi1 = bias ? bias[n1] : 0.0f;
#pragma unroll
    for (int i = 0; i < 8; ++i) {
        const int m = mBase + i + half * 8;   // C/D layout: VGPR i -> M = i + 8*half
        OUT[(size_t)m * Kout + n0] = acc0[i] + bi0;
        OUT[(size_t)m * Kout + n1] = acc1[i] + bi1;
    }
}

// ------------------------- helper / elementwise kernels ---------------------

__global__ void k_cvt_transpose(const float* __restrict__ W,
                                __hip_bfloat16* __restrict__ WT,
                                int Kin, int Kout)
{
    int idx = blockIdx.x * blockDim.x + threadIdx.x;
    if (idx >= Kin * Kout) return;
    int i = idx / Kout, o = idx - i * Kout;
    WT[(size_t)o * Kin + i] = __float2bfloat16(W[idx]);
}

__global__ void k_cvt_bf16(const float* __restrict__ X,
                           __hip_bfloat16* __restrict__ Y, int n)
{
    int idx = blockIdx.x * blockDim.x + threadIdx.x;
    if (idx < n) Y[idx] = __float2bfloat16(X[idx]);
}

__global__ void k_bn_stats(const float* __restrict__ x,
                           float* __restrict__ mu, float* __restrict__ rstd)
{
    const int c = blockIdx.x;
    __shared__ float ssum[256];
    __shared__ float ssq[256];
    float s = 0.f, s2 = 0.f;
    for (int i = threadIdx.x; i < NN; i += 256) {
        float v = x[(size_t)i * CC + c];
        s += v; s2 += v * v;
    }
    ssum[threadIdx.x] = s; ssq[threadIdx.x] = s2;
    __syncthreads();
    for (int st = 128; st > 0; st >>= 1) {
        if (threadIdx.x < st) {
            ssum[threadIdx.x] += ssum[threadIdx.x + st];
            ssq[threadIdx.x]  += ssq[threadIdx.x + st];
        }
        __syncthreads();
    }
    if (threadIdx.x == 0) {
        float m = ssum[0] / (float)NN;
        float var = ssq[0] / (float)NN - m * m;
        mu[c] = m;
        rstd[c] = rsqrtf(var + 1e-5f);
    }
}

__global__ void k_bn_apply(const float* __restrict__ x,
                           const float* __restrict__ mu, const float* __restrict__ rstd,
                           const float* __restrict__ g,  const float* __restrict__ b,
                           float* __restrict__ y, __hip_bfloat16* __restrict__ ybf)
{
    int idx = blockIdx.x * blockDim.x + threadIdx.x;
    if (idx >= NN * CC) return;
    int c = idx & (CC - 1);
    float v = (x[idx] - mu[c]) * rstd[c] * g[c] + b[c];
    y[idx] = v;
    ybf[idx] = __float2bfloat16(v);
}

__global__ void k_deg_init(float* deg) {
    int i = blockIdx.x * blockDim.x + threadIdx.x;
    if (i < NN) deg[i] = 1.0f;                       // self-loop
}
__global__ void k_deg_scatter(const int* __restrict__ dst, float* deg) {
    int e = blockIdx.x * blockDim.x + threadIdx.x;
    if (e < NE) atomicAdd(&deg[dst[e]], 1.0f);
}
__global__ void k_deg_fin(float* deg) {
    int i = blockIdx.x * blockDim.x + threadIdx.x;
    if (i < NN) deg[i] = rsqrtf(fmaxf(deg[i], 1e-12f));   // -> dinv in place
}

__global__ void k_gcn_init(const float* __restrict__ xw, const float* __restrict__ dinv,
                           const float* __restrict__ b, float* __restrict__ out)
{
    int idx = blockIdx.x * blockDim.x + threadIdx.x;
    if (idx >= NN * CC) return;
    int i = idx >> 7, c = idx & 127;
    float di = dinv[i];
    out[idx] = xw[idx] * di * di + b[c];
}

__global__ void k_gcn_scatter(const float* __restrict__ xw, const float* __restrict__ dinv,
                              const int* __restrict__ src, const int* __restrict__ dst,
                              float* __restrict__ out)
{
    int idx = blockIdx.x * blockDim.x + threadIdx.x;
    if (idx >= NE * CC) return;
    int e = idx >> 7, c = idx & 127;
    int s = src[e], d = dst[e];
    atomicAdd(&out[(size_t)d * CC + c],
              xw[(size_t)s * CC + c] * dinv[s] * dinv[d]);
}

// order-preserving float<->uint for atomicMax-based segment max
__device__ __forceinline__ unsigned f2ord(float f) {
    unsigned u = __float_as_uint(f);
    return (u & 0x80000000u) ? ~u : (u | 0x80000000u);
}
__device__ __forceinline__ float ord2f(unsigned e) {
    return __uint_as_float((e & 0x80000000u) ? (e ^ 0x80000000u) : ~e);
}

__global__ void k_att_init(unsigned* menc, float* s) {
    int i = blockIdx.x * blockDim.x + threadIdx.x;
    if (i < NN * HH) { menc[i] = 0x007fffffu /* enc(-inf) */; s[i] = 0.f; }
}

__global__ void k_att_alpha(const float* __restrict__ q, const float* __restrict__ k,
                            const int* __restrict__ src, const int* __restrict__ dst,
                            float* __restrict__ alpha, unsigned* __restrict__ menc)
{
    int idx = blockIdx.x * blockDim.x + threadIdx.x;
    if (idx >= NE * HH) return;
    int e = idx >> 2, h = idx & 3;
    int d = dst[e], sn = src[e];
    const float4* qa = (const float4*)(q + (size_t)d  * HCC + h * CC);
    const float4* ka = (const float4*)(k + (size_t)sn * HCC + h * CC);
    float acc = 0.f;
#pragma unroll 8
    for (int j = 0; j < 32; ++j) {
        float4 a = qa[j], b = ka[j];
        acc += a.x * b.x + a.y * b.y + a.z * b.z + a.w * b.w;
    }
    acc *= 0.08838834764831845f;    // 1/sqrt(128)
    alpha[idx] = acc;
    atomicMax(&menc[(size_t)d * HH + h], f2ord(acc));
}

__global__ void k_att_exp(float* __restrict__ alpha, const unsigned* __restrict__ menc,
                          float* __restrict__ s, const int* __restrict__ dst)
{
    int idx = blockIdx.x * blockDim.x + threadIdx.x;
    if (idx >= NE * HH) return;
    int e = idx >> 2, h = idx & 3;
    int d = dst[e];
    float ee = expf(alpha[idx] - ord2f(menc[(size_t)d * HH + h]));
    alpha[idx] = ee;
    atomicAdd(&s[(size_t)d * HH + h], ee);
}

__global__ void k_att_scatter(const float* __restrict__ v, const float* __restrict__ alpha,
                              const float* __restrict__ s,
                              const int* __restrict__ src, const int* __restrict__ dst,
                              float* __restrict__ out)
{
    int idx = blockIdx.x * blockDim.x + threadIdx.x;
    if (idx >= NE * HCC) return;
    int e = idx >> 9, j = idx & 511;
    int h = j >> 7;
    int d = dst[e], sn = src[e];
    float a = alpha[(size_t)e * HH + h] / fmaxf(s[(size_t)d * HH + h], 1e-16f);
    atomicAdd(&out[(size_t)d * CC + (j & 127)],
              v[(size_t)sn * HCC + j] * a * 0.25f);   // mean over 4 heads
}

// LayerNorm (in place) + bf16 copy; one wave32 per node, 1 float4 per lane.
__global__ void k_layernorm(float* __restrict__ x,
                            const float* __restrict__ g, const float* __restrict__ b,
                            __hip_bfloat16* __restrict__ xbf)
{
    int node = blockIdx.x * blockDim.y + threadIdx.y;
    if (node >= NN) return;
    int lane = threadIdx.x;
    float4* row = (float4*)(x + (size_t)node * CC);
    float4 v = row[lane];
    float sum = v.x + v.y + v.z + v.w;
#pragma unroll
    for (int o = 16; o > 0; o >>= 1) sum += __shfl_xor(sum, o, 32);
    float mu = sum * (1.0f / CC);
    float d0 = v.x - mu, d1 = v.y - mu, d2 = v.z - mu, d3 = v.w - mu;
    float sq = d0 * d0 + d1 * d1 + d2 * d2 + d3 * d3;
#pragma unroll
    for (int o = 16; o > 0; o >>= 1) sq += __shfl_xor(sq, o, 32);
    float rstd = rsqrtf(sq * (1.0f / CC) + 1e-5f);
    float4 gg = ((const float4*)g)[lane];
    float4 bb = ((const float4*)b)[lane];
    float4 o4;
    o4.x = d0 * rstd * gg.x + bb.x;
    o4.y = d1 * rstd * gg.y + bb.y;
    o4.z = d2 * rstd * gg.z + bb.z;
    o4.w = d3 * rstd * gg.w + bb.w;
    row[lane] = o4;
    __hip_bfloat16* bp = xbf + (size_t)node * CC + lane * 4;
    bp[0] = __float2bfloat16(o4.x); bp[1] = __float2bfloat16(o4.y);
    bp[2] = __float2bfloat16(o4.z); bp[3] = __float2bfloat16(o4.w);
}

__global__ void k_finalize(float* __restrict__ x, __hip_bfloat16* __restrict__ xbf,
                           int n, int do_silu)
{
    int idx = blockIdx.x * blockDim.x + threadIdx.x;
    if (idx >= n) return;
    float v = x[idx];
    if (do_silu) { v = v / (1.0f + expf(-v)); x[idx] = v; }
    xbf[idx] = __float2bfloat16(v);
}

__global__ void k_residual(const float* __restrict__ x0, const float* __restrict__ xf,
                           float* __restrict__ out, int n)
{
    int idx = blockIdx.x * blockDim.x + threadIdx.x;
    if (idx < n) out[idx] = x0[idx] + xf[idx];
}

// ---------------------------------------------------------------------------

extern "C" void kernel_launch(void* const* d_in, const int* in_sizes, int n_in,
                              void* d_out, int out_size, void* d_ws, size_t ws_size,
                              hipStream_t stream)
{
    const float* x    = (const float*)d_in[0];
    const int*   ei   = (const int*)  d_in[1];
    const float* cond = (const float*)d_in[2];
    const float* bn_g = (const float*)d_in[3];
    const float* bn_b = (const float*)d_in[4];
    const float* Wpi  = (const float*)d_in[5];  const float* bpi = (const float*)d_in[6];
    const float* Wq1  = (const float*)d_in[7];  const float* bq1 = (const float*)d_in[8];
    const float* Wk1  = (const float*)d_in[9];  const float* bk1 = (const float*)d_in[10];
    const float* Wv1  = (const float*)d_in[11]; const float* bv1 = (const float*)d_in[12];
    const float* Ws1  = (const float*)d_in[13]; const float* bs1 = (const float*)d_in[14];
    const float* lng1 = (const float*)d_in[15]; const float* lnb1 = (const float*)d_in[16];
    const float* Wq2  = (const float*)d_in[17]; const float* bq2 = (const float*)d_in[18];
    const float* Wk2  = (const float*)d_in[19]; const float* bk2 = (const float*)d_in[20];
    const float* Wv2  = (const float*)d_in[21]; const float* bv2 = (const float*)d_in[22];
    const float* Ws2  = (const float*)d_in[23]; const float* bs2 = (const float*)d_in[24];
    const float* lng2 = (const float*)d_in[25]; const float* lnb2 = (const float*)d_in[26];
    const float* Wg1  = (const float*)d_in[27]; const float* bg1 = (const float*)d_in[28];
    const float* Wg2  = (const float*)d_in[29]; const float* bg2 = (const float*)d_in[30];
    const float* Wpo  = (const float*)d_in[31]; const float* bpo = (const float*)d_in[32];
    const int* src = ei;
    const int* dst = ei + NE;
    float* out = (float*)d_out;

    // ---- workspace carve-up -------------------------------------------------
    size_t off = 0;
    auto alloc = [&](size_t bytes) -> void* {
        void* p = (void*)((char*)d_ws + off);
        off = (off + bytes + 255) & ~(size_t)255;
        return p;
    };
    float* buf0 = (float*)alloc((size_t)NN * CC * 4);
    float* buf1 = (float*)alloc((size_t)NN * CC * 4);
    float* buf2 = (float*)alloc((size_t)NN * CC * 4);
    float* big1 = (float*)alloc((size_t)NN * HCC * 4);   // q
    float* big2 = (float*)alloc((size_t)NN * HCC * 4);   // k, then v
    __hip_bfloat16* curbf  = (__hip_bfloat16*)alloc((size_t)NN * CC * 2);
    __hip_bfloat16* condbf = (__hip_bfloat16*)alloc((size_t)NN * DD * 2);
    float*    dinv  = (float*)alloc((size_t)NN * 4);
    float*    mu    = (float*)alloc(CC * 4);
    float*    rstd  = (float*)alloc(CC * 4);
    unsigned* menc  = (unsigned*)alloc((size_t)NN * HH * 4);
    float*    ssum  = (float*)alloc((size_t)NN * HH * 4);
    float*    alpha = (float*)alloc((size_t)NE * HH * 4);
    __hip_bfloat16* WpiT = (__hip_bfloat16*)alloc((size_t)CC * CC * 2);
    __hip_bfloat16* Wq1T = (__hip_bfloat16*)alloc((size_t)CC * HCC * 2);
    __hip_bfloat16* Wk1T = (__hip_bfloat16*)alloc((size_t)DD * HCC * 2);
    __hip_bfloat16* Wv1T = (__hip_bfloat16*)alloc((size_t)DD * HCC * 2);
    __hip_bfloat16* Ws1T = (__hip_bfloat16*)alloc((size_t)CC * CC * 2);
    __hip_bfloat16* Wq2T = (__hip_bfloat16*)alloc((size_t)CC * HCC * 2);
    __hip_bfloat16* Wk2T = (__hip_bfloat16*)alloc((size_t)DD * HCC * 2);
    __hip_bfloat16* Wv2T = (__hip_bfloat16*)alloc((size_t)DD * HCC * 2);
    __hip_bfloat16* Ws2T = (__hip_bfloat16*)alloc((size_t)CC * CC * 2);
    __hip_bfloat16* Wg1T = (__hip_bfloat16*)alloc((size_t)CC * CC * 2);
    __hip_bfloat16* Wg2T = (__hip_bfloat16*)alloc((size_t)CC * CC * 2);
    __hip_bfloat16* WpoT = (__hip_bfloat16*)alloc((size_t)CC * CC * 2);

    auto g1d = [](long long total) { return dim3((unsigned)((total + 255) / 256)); };
    const dim3 B256(256);

    auto cvtT = [&](const float* W, __hip_bfloat16* WT, int Kin, int Kout) {
        k_cvt_transpose<<<g1d((long long)Kin * Kout), B256, 0, stream>>>(W, WT, Kin, Kout);
    };
    auto gemm = [&](const __hip_bfloat16* A, const __hip_bfloat16* WT,
                    const float* bias, float* O, int Kin, int Kout) {
        dim3 grid(NN / 16, Kout / 128);
        dim3 block(32, 4);
        k_gemm_bf16<<<grid, block, 0, stream>>>(A, WT, bias, O, Kin, Kout);
    };
    auto gcn = [&](const __hip_bfloat16* in_bf, const __hip_bfloat16* WT,
                   const float* b, float* tmp, float* o) {
        gemm(in_bf, WT, nullptr, tmp, CC, CC);
        k_gcn_init<<<g1d((long long)NN * CC), B256, 0, stream>>>(tmp, dinv, b, o);
        k_gcn_scatter<<<g1d((long long)NE * CC), B256, 0, stream>>>(tmp, dinv, src, dst, o);
    };

    // ---- weight conversion (fp32 -> bf16, transposed) ----------------------
    cvtT(Wpi, WpiT, CC, CC);
    cvtT(Wq1, Wq1T, CC, HCC);  cvtT(Wk1, Wk1T, DD, HCC);
    cvtT(Wv1, Wv1T, DD, HCC);  cvtT(Ws1, Ws1T, CC, CC);
    cvtT(Wq2, Wq2T, CC, HCC);  cvtT(Wk2, Wk2T, DD, HCC);
    cvtT(Wv2, Wv2T, DD, HCC);  cvtT(Ws2, Ws2T, CC, CC);
    cvtT(Wg1, Wg1T, CC, CC);   cvtT(Wg2, Wg2T, CC, CC);
    cvtT(Wpo, WpoT, CC, CC);
    k_cvt_bf16<<<g1d((long long)NN * DD), B256, 0, stream>>>(cond, condbf, NN * DD);

    // ---- BatchNorm ----------------------------------------------------------
    k_bn_stats<<<dim3(CC), B256, 0, stream>>>(x, mu, rstd);
    k_bn_apply<<<g1d((long long)NN * CC), B256, 0, stream>>>(x, mu, rstd, bn_g, bn_b, buf0, curbf);

    // ---- degrees ------------------------------------------------------------
    k_deg_init<<<g1d(NN), B256, 0, stream>>>(dinv);
    k_deg_scatter<<<g1d(NE), B256, 0, stream>>>(dst, dinv);
    k_deg_fin<<<g1d(NN), B256, 0, stream>>>(dinv);

    // ---- GCN (pi): cur = buf2 ----------------------------------------------
    gcn(curbf, WpiT, bpi, buf1, buf2);
    k_finalize<<<g1d((long long)NN * CC), B256, 0, stream>>>(buf2, curbf, NN * CC, 0);

    // ---- two infusion (attention) layers ------------------------------------
    const __hip_bfloat16* WqT[2] = {Wq1T, Wq2T};
    const __hip_bfloat16* WkT[2] = {Wk1T, Wk2T};
    const __hip_bfloat16* WvT[2] = {Wv1T, Wv2T};
    const __hip_bfloat16* WsT[2] = {Ws1T, Ws2T};
    const float* bqA[2] = {bq1, bq2}; const float* bkA[2] = {bk1, bk2};
    const float* bvA[2] = {bv1, bv2}; const float* bsA[2] = {bs1, bs2};
    const float* lngA[2] = {lng1, lng2}; const float* lnbA[2] = {lnb1, lnb2};
    float* skipbuf[2] = {buf0, buf1};    // layer1 accumulates into buf0, layer2 into buf1

    for (int l = 0; l < 2; ++l) {
        gemm(curbf, WqT[l], bqA[l], big1, CC, HCC);          // q  [N,512]
        gemm(condbf, WkT[l], bkA[l], big2, DD, HCC);         // k  [N,512]
        k_att_init<<<g1d((long long)NN * HH), B256, 0, stream>>>(menc, ssum);
        k_att_alpha<<<g1d((long long)NE * HH), B256, 0, stream>>>(big1, big2, src, dst, alpha, menc);
        k_att_exp<<<g1d((long long)NE * HH), B256, 0, stream>>>(alpha, menc, ssum, dst);
        gemm(curbf, WsT[l], bsA[l], skipbuf[l], CC, CC);     // skip (accumulator init)
        gemm(condbf, WvT[l], bvA[l], big2, DD, HCC);         // v overwrites k
        k_att_scatter<<<g1d((long long)NE * HCC), B256, 0, stream>>>(big2, alpha, ssum,
                                                                     src, dst, skipbuf[l]);
        k_layernorm<<<dim3((NN + 7) / 8), dim3(32, 8), 0, stream>>>(skipbuf[l], lngA[l],
                                                                    lnbA[l], curbf);
        // cur = skipbuf[l]
    }

    // ---- GCN g1 (+SiLU): cur(buf1) -> buf0 ----------------------------------
    gcn(curbf, Wg1T, bg1, buf2, buf0);
    k_finalize<<<g1d((long long)NN * CC), B256, 0, stream>>>(buf0, curbf, NN * CC, 1);

    // ---- GCN g2: cur(buf0) -> buf1 ------------------------------------------
    gcn(curbf, Wg2T, bg2, buf2, buf1);
    k_finalize<<<g1d((long long)NN * CC), B256, 0, stream>>>(buf1, curbf, NN * CC, 0);

    // ---- GCN po: cur(buf1) -> buf0, then residual ---------------------------
    gcn(curbf, WpoT, bpo, buf2, buf0);
    k_residual<<<g1d((long long)NN * CC), B256, 0, stream>>>(x, buf0, out, NN * CC);

    (void)in_sizes; (void)n_in; (void)out_size; (void)ws_size;
}
